// SpiralAttentionBlock_84885733638498
// MI455X (gfx1250) — matrix operand order repaired and checked by reading, not compiled, not run on hardware
//
#include <hip/hip_runtime.h>
#include <stdint.h>

#define CCH 256
#define HEADS 8
#define HDIM 32
#define NTOK 4096
#define NGRP 32
#define EPSV 1e-5f

typedef __attribute__((ext_vector_type(16))) __bf16 v16bf;
typedef __attribute__((ext_vector_type(8)))  float  v8f;

union BF16x16 { v16bf v; uint16_t u[16]; uint32_t d[8]; };

__device__ __forceinline__ uint16_t f2bf(float f) {
    uint32_t b = __float_as_uint(f);
    uint32_t r = b + 0x7FFFu + ((b >> 16) & 1u);   // round-to-nearest-even
    return (uint16_t)(r >> 16);
}

// ---------------- GroupNorm statistics: one block per group ----------------
__global__ void gn_stats(const float* __restrict__ x, float* __restrict__ stats) {
    __shared__ float ss[256], sq[256];
    int g = blockIdx.x, t = threadIdx.x;
    const float* xp = x + (size_t)g * 8 * NTOK;
    float s = 0.f, q = 0.f;
    for (int i = t; i < 8 * NTOK; i += 256) { float v = xp[i]; s += v; q += v * v; }
    ss[t] = s; sq[t] = q; __syncthreads();
    for (int o = 128; o > 0; o >>= 1) {
        if (t < o) { ss[t] += ss[t + o]; sq[t] += sq[t + o]; }
        __syncthreads();
    }
    if (t == 0) {
        float mu = ss[0] * (1.f / (8 * NTOK));
        float var = sq[0] * (1.f / (8 * NTOK)) - mu * mu;
        stats[g * 2] = mu;
        stats[g * 2 + 1] = rsqrtf(var + EPSV);
    }
}

// ------- normalize + affine + cast to bf16, write transposed hT[n][c] -------
__global__ void gn_norm(const float* __restrict__ x, const float* __restrict__ stats,
                        const float* __restrict__ nw, const float* __restrict__ nb,
                        uint16_t* __restrict__ hT) {
    int idx = blockIdx.x * 256 + threadIdx.x;   // idx = n*C + c
    int c = idx & (CCH - 1);
    int n = idx >> 8;
    int g = c >> 3;
    float v = (x[(size_t)c * NTOK + n] - stats[g * 2]) * stats[g * 2 + 1] * nw[c] + nb[c];
    hT[idx] = f2bf(v);
}

// ---------------- weight fp32 -> bf16 ----------------
__global__ void wconv(const float* __restrict__ qw, const float* __restrict__ pw,
                      uint16_t* __restrict__ qwb, uint16_t* __restrict__ pwb) {
    int i = blockIdx.x * 256 + threadIdx.x;
    if (i < 3 * CCH * CCH) qwb[i] = f2bf(qw[i]);
    else { int j = i - 3 * CCH * CCH; pwb[j] = f2bf(pw[j]); }
}

// ---------------- spiral bias score vector: sb[n] = 0.1*mean_h(...) ----------------
__global__ void spiral(const float* __restrict__ cc, const float* __restrict__ str,
                       const float* __restrict__ freq, float* __restrict__ sb) {
    int n = blockIdx.x * 256 + threadIdx.x;    // 4096
    int t = n >> 10, rem = n & 1023, y = rem >> 5, xc = rem & 31;
    float gy = y * (1.f / 31.f), gx = xc * (1.f / 31.f);
    float dy = gy - cc[t * 2 + 0];
    float dx = gx - cc[t * 2 + 1];
    float r = sqrtf(dx * dx + dy * dy);
    float th = atan2f(dy, dx);
    float a = 0.f;
#pragma unroll
    for (int h = 0; h < HEADS; h++) a += str[h] * __cosf(th - freq[h] * r);
    sb[n] = 0.1f * a * (1.f / HEADS);
}

// ---------------- QKV GEMM: qkv[o][n] = sum_c w[o][c]*hT[n][c], bf16 WMMA ----------------
__global__ void __launch_bounds__(128) qkv_gemm(const uint16_t* __restrict__ wb,
                                                const uint16_t* __restrict__ hT,
                                                const float* __restrict__ qkvb,
                                                uint16_t* __restrict__ out) {
    int wave = (blockIdx.x * blockDim.x + threadIdx.x) >> 5;
    int lane = threadIdx.x & 31;
    int to = wave % 48, tn = wave / 48;
    int o0 = to * 16, n0 = tn * 16;
    int m = lane & 15, lh = lane >> 4;
    v8f acc = {};
    const uint16_t* wrow = wb + (size_t)(o0 + m) * CCH;   // A: row = output channel
    const uint16_t* hrow = hT + (size_t)(n0 + m) * CCH;   // B: col = token
    for (int c0 = 0; c0 < CCH; c0 += 32) {
        BF16x16 A, B;
        int ka = c0 + lh * 8;
        *(uint4*)&A.d[0] = *(const uint4*)(wrow + ka);
        *(uint4*)&A.d[4] = *(const uint4*)(wrow + ka + 16);
        int kb = c0 + lh * 8;
        *(uint4*)&B.d[0] = *(const uint4*)(hrow + kb);
        *(uint4*)&B.d[4] = *(const uint4*)(hrow + kb + 16);
        acc = __builtin_amdgcn_wmma_f32_16x16x32_bf16(false, A.v, false, B.v,
                                                      (short)0, acc, false, false);
    }
    const float scale = 0.17677669529663687f; // 32^-0.5, folded into q
#pragma unroll
    for (int r = 0; r < 8; r++) {
        int orow = o0 + r + lh * 8;
        float v = acc[r] + qkvb[orow];
        if (orow < CCH) v *= scale;
        out[(size_t)orow * NTOK + n0 + m] = f2bf(v);
    }
}

// ---------------- transpose q,k to [h][n][d] for contiguous WMMA operand loads ----------------
__global__ void qk_transpose(const uint16_t* __restrict__ qkv,
                             uint16_t* __restrict__ Qt, uint16_t* __restrict__ Kt) {
    int idx = blockIdx.x * 256 + threadIdx.x;       // 2 * 8 * 4096 * 32
    int which = idx >> 20;
    int rem = idx & 0xFFFFF;
    int d = rem & 31;
    int n = (rem >> 5) & 4095;
    int h = rem >> 17;
    uint16_t v = qkv[(size_t)(which * CCH + h * HDIM + d) * NTOK + n];
    (which ? Kt : Qt)[((size_t)h * NTOK + n) * HDIM + d] = v;
}

// ---- flash attention: one wave = 32 queries (2 row-tiles) of one head ----
// K/V B-operands are loaded once per 32-key step and shared by both row-tiles:
// 8 WMMAs per step against the same 128B/lane of K/V traffic.
__global__ void __launch_bounds__(64) flash_attn(const uint16_t* __restrict__ Qt,
                                                 const uint16_t* __restrict__ Kt,
                                                 const uint16_t* __restrict__ qkv,
                                                 const float* __restrict__ sb,
                                                 uint16_t* __restrict__ attnT) {
    __shared__ __align__(16) uint16_t pb[2][2][16][32];
    int wslot = threadIdx.x >> 5;
    int lane  = threadIdx.x & 31;
    int wave  = blockIdx.x * 2 + wslot;     // 8 heads * 128 query-pair tiles
    int h   = wave >> 7;
    int qt  = wave & 127;
    int nq0 = qt * 32;
    int m = lane & 15, lh = lane >> 4;

    // persistent Q A-operands (A layout: lane = row, lanes 16-31 hold K=8..15,24..31)
    BF16x16 QA[2];
#pragma unroll
    for (int t = 0; t < 2; t++) {
        const uint16_t* qrow = Qt + ((size_t)h * NTOK + nq0 + t * 16 + m) * HDIM;
        *(uint4*)&QA[t].d[0] = *(const uint4*)(qrow + lh * 8);
        *(uint4*)&QA[t].d[4] = *(const uint4*)(qrow + lh * 8 + 16);
    }

    float bq[2][8];
#pragma unroll
    for (int t = 0; t < 2; t++)
#pragma unroll
        for (int r = 0; r < 8; r++) bq[t][r] = sb[nq0 + t * 16 + r + lh * 8];

    float mrow[2][8], lrow[2][8];
#pragma unroll
    for (int t = 0; t < 2; t++)
#pragma unroll
        for (int r = 0; r < 8; r++) { mrow[t][r] = -3.0e38f; lrow[t][r] = 0.f; }
    v8f o[2][2] = {};

    const uint16_t* vbase = qkv + (size_t)(2 * CCH + h * HDIM) * NTOK;

    for (int k0 = 0; k0 < NTOK; k0 += 32) {
        // K^T B-operands: B layout lane = key column, lanes 16-31 hold K(d)=8..15,24..31
        BF16x16 KB[2];
        const uint16_t* kr0 = Kt + ((size_t)h * NTOK + k0 + m) * HDIM + lh * 8;
#pragma unroll
        for (int j = 0; j < 2; j++) {
            const uint16_t* kr = kr0 + j * 16 * HDIM;
            *(uint4*)&KB[j].d[0] = *(const uint4*)(kr);
            *(uint4*)&KB[j].d[4] = *(const uint4*)(kr + 16);
        }

        v8f s[2][2] = {};
#pragma unroll
        for (int t = 0; t < 2; t++)
#pragma unroll
            for (int j = 0; j < 2; j++)
                s[t][j] = __builtin_amdgcn_wmma_f32_16x16x32_bf16(
                    false, QA[t].v, false, KB[j].v, (short)0, s[t][j], false, false);

        float bk0 = sb[k0 + m], bk1 = sb[k0 + 16 + m];

        // online softmax per row-tile, write P (bf16) into LDS for transpose
#pragma unroll
        for (int t = 0; t < 2; t++) {
#pragma unroll
            for (int r = 0; r < 8; r++) {
                float t0 = s[t][0][r] + bq[t][r] + bk0;
                float t1 = s[t][1][r] + bq[t][r] + bk1;
                float loc = fmaxf(t0, t1);
#pragma unroll
                for (int off = 1; off < 16; off <<= 1)
                    loc = fmaxf(loc, __shfl_xor(loc, off, 32));
                float mn = fmaxf(mrow[t][r], loc);
                float e0 = __expf(t0 - mn), e1 = __expf(t1 - mn);
                float rs = e0 + e1;
#pragma unroll
                for (int off = 1; off < 16; off <<= 1)
                    rs += __shfl_xor(rs, off, 32);
                float alpha = __expf(mrow[t][r] - mn);
                lrow[t][r] = lrow[t][r] * alpha + rs;
                mrow[t][r] = mn;
                o[t][0][r] *= alpha; o[t][1][r] *= alpha;
                pb[wslot][t][r + lh * 8][m]      = f2bf(e0);
                pb[wslot][t][r + lh * 8][m + 16] = f2bf(e1);
            }
        }
        asm volatile("s_wait_dscnt 0" ::: "memory");

        // P tiles back out in A layout
        BF16x16 PA[2];
#pragma unroll
        for (int t = 0; t < 2; t++) {
            *(uint4*)&PA[t].d[0] = *(const uint4*)&pb[wslot][t][m][lh * 8];
            *(uint4*)&PA[t].d[4] = *(const uint4*)&pb[wslot][t][m][lh * 8 + 16];
        }

        // V B-operands: col = head-dim, contiguous keys per lane (shared by both row-tiles)
        BF16x16 VB[2];
        const uint16_t* vr0 = vbase + (size_t)m * NTOK + k0 + lh * 8;
#pragma unroll
        for (int j = 0; j < 2; j++) {
            const uint16_t* vr = vr0 + j * 16 * NTOK;
            *(uint4*)&VB[j].d[0] = *(const uint4*)(vr);
            *(uint4*)&VB[j].d[4] = *(const uint4*)(vr + 16);
        }

#pragma unroll
        for (int t = 0; t < 2; t++)
#pragma unroll
            for (int j = 0; j < 2; j++)
                o[t][j] = __builtin_amdgcn_wmma_f32_16x16x32_bf16(
                    false, PA[t].v, false, VB[j].v, (short)0, o[t][j], false, false);

        if (k0 + 32 < NTOK) __builtin_prefetch(kr0 + 32 * HDIM, 0, 1);
    }

#pragma unroll
    for (int t = 0; t < 2; t++)
#pragma unroll
        for (int r = 0; r < 8; r++) {
            int nq = nq0 + t * 16 + r + lh * 8;
            float inv = 1.f / lrow[t][r];
            attnT[(size_t)nq * CCH + h * HDIM + m]      = f2bf(o[t][0][r] * inv);
            attnT[(size_t)nq * CCH + h * HDIM + 16 + m] = f2bf(o[t][1][r] * inv);
        }
}

// ---------------- proj GEMM + bias + residual (fp32 out) ----------------
__global__ void __launch_bounds__(128) proj_gemm(const uint16_t* __restrict__ pwb,
                                                 const uint16_t* __restrict__ attnT,
                                                 const float* __restrict__ pbias,
                                                 const float* __restrict__ x,
                                                 float* __restrict__ out) {
    int wave = (blockIdx.x * blockDim.x + threadIdx.x) >> 5;
    int lane = threadIdx.x & 31;
    int to = wave % 16, tn = wave / 16;
    int o0 = to * 16, n0 = tn * 16;
    int m = lane & 15, lh = lane >> 4;
    v8f acc = {};
    const uint16_t* wrow = pwb   + (size_t)(o0 + m) * CCH;
    const uint16_t* arow = attnT + (size_t)(n0 + m) * CCH;
    for (int c0 = 0; c0 < CCH; c0 += 32) {
        BF16x16 A, B;
        *(uint4*)&A.d[0] = *(const uint4*)(wrow + c0 + lh * 8);
        *(uint4*)&A.d[4] = *(const uint4*)(wrow + c0 + lh * 8 + 16);
        *(uint4*)&B.d[0] = *(const uint4*)(arow + c0 + lh * 8);
        *(uint4*)&B.d[4] = *(const uint4*)(arow + c0 + lh * 8 + 16);
        acc = __builtin_amdgcn_wmma_f32_16x16x32_bf16(false, A.v, false, B.v,
                                                      (short)0, acc, false, false);
    }
#pragma unroll
    for (int r = 0; r < 8; r++) {
        int orow = o0 + r + lh * 8;
        int n = n0 + m;
        out[(size_t)orow * NTOK + n] = acc[r] + pbias[orow] + x[(size_t)orow * NTOK + n];
    }
}

extern "C" void kernel_launch(void* const* d_in, const int* in_sizes, int n_in,
                              void* d_out, int out_size, void* d_ws, size_t ws_size,
                              hipStream_t stream) {
    const float* x     = (const float*)d_in[0];
    const float* cc    = (const float*)d_in[1];
    const float* nw    = (const float*)d_in[2];
    const float* nb    = (const float*)d_in[3];
    const float* qw    = (const float*)d_in[4];
    const float* qb    = (const float*)d_in[5];
    const float* pw    = (const float*)d_in[6];
    const float* pbias = (const float*)d_in[7];
    const float* str   = (const float*)d_in[8];
    const float* freq  = (const float*)d_in[9];
    float* out = (float*)d_out;

    char* ws = (char*)d_ws;
    size_t off = 0;
    auto alloc = [&](size_t bytes) { char* p = ws + off; off = (off + bytes + 255) & ~(size_t)255; return p; };
    float*    stats = (float*)   alloc(NGRP * 2 * sizeof(float));
    float*    sb    = (float*)   alloc(NTOK * sizeof(float));
    uint16_t* qwb   = (uint16_t*)alloc((size_t)3 * CCH * CCH * 2);
    uint16_t* pwb   = (uint16_t*)alloc((size_t)CCH * CCH * 2);
    uint16_t* hT    = (uint16_t*)alloc((size_t)NTOK * CCH * 2);
    uint16_t* qkv   = (uint16_t*)alloc((size_t)3 * CCH * NTOK * 2);
    uint16_t* Qt    = (uint16_t*)alloc((size_t)HEADS * NTOK * HDIM * 2);
    uint16_t* Kt    = (uint16_t*)alloc((size_t)HEADS * NTOK * HDIM * 2);
    uint16_t* attnT = (uint16_t*)alloc((size_t)NTOK * CCH * 2);
    (void)ws_size; (void)n_in; (void)in_sizes; (void)out_size;

    gn_stats<<<NGRP, 256, 0, stream>>>(x, stats);
    gn_norm<<<(NTOK * CCH) / 256, 256, 0, stream>>>(x, stats, nw, nb, hT);
    wconv<<<(4 * CCH * CCH) / 256, 256, 0, stream>>>(qw, pw, qwb, pwb);
    spiral<<<NTOK / 256, 256, 0, stream>>>(cc, str, freq, sb);
    qkv_gemm<<<(48 * 256) / 4, 128, 0, stream>>>(qwb, hT, qb, qkv);
    qk_transpose<<<(2 * HEADS * NTOK * HDIM) / 256, 256, 0, stream>>>(qkv, Qt, Kt);
    flash_attn<<<(HEADS * 128) / 2, 64, 0, stream>>>(Qt, Kt, qkv, sb, attnT);
    proj_gemm<<<(16 * 256) / 4, 128, 0, stream>>>(pwb, attnT, pbias, x, out);
}
